// Meta_Transformer_Layer_7499012899506
// MI455X (gfx1250) — matrix operand order repaired and checked by reading, not compiled, not run on hardware
//
#include <hip/hip_runtime.h>
#include <hip/hip_bf16.h>
#include <math.h>

// ---------------------------------------------------------------------------
// Meta transformer layer for MI455X (gfx1250, wave32, WMMA).
// GEMMs: v_wmma_f32_16x16x32_f16, f32 accumulation, 2x4 register blocking per
// wave (32x64 C slab), 128x64x64 block tiles. Attention: flash-style online
// softmax, f32 statistics. LayerNorm in f32.
// ---------------------------------------------------------------------------

typedef __attribute__((ext_vector_type(16))) _Float16 v16h;
typedef __attribute__((ext_vector_type(8)))  _Float16 v8h;
typedef __attribute__((ext_vector_type(8)))  float    v8f;

#define Bc 8
#define Sc 1024
#define Ec 1024
#define Hc 16
#define Dc 64
#define Mc (Bc*Sc)        // 8192 rows

#define LDB 72            // GEMM LDS row stride (64 + 8 pad halfs, 16B-aligned chunks)
#define LKS 72            // attn K-tile stride (64 + 8)
#define LVT 40            // attn Vt stride (32 + 8)
#define LPB 40            // attn P stride  (32 + 8)

__device__ __forceinline__ v16h make_v16h(v8h lo, v8h hi) {
  return __builtin_shufflevector(lo, hi, 0,1,2,3,4,5,6,7,8,9,10,11,12,13,14,15);
}

__device__ __forceinline__ v8f wmma_f16(v16h a, v16h b, v8f c) {
  // (neg_a, A, neg_b, B, c_mod, C, reuse_a, reuse_b)
  return __builtin_amdgcn_wmma_f32_16x16x32_f16(false, a, false, b, (short)0, c, false, false);
}

// A-fragment (16x32 f16): lane l -> row base+l%16, K run 16*(l/16)..+15
__device__ __forceinline__ v16h load_afrag(const _Float16* row, int hf) {
  v8h lo = *(const v8h*)(row + 16*hf);
  v8h hi = *(const v8h*)(row + 16*hf + 8);
  return make_v16h(lo, hi);
}

// B-fragment (32x16 f16) from N-major tile: lane l -> col l%16, K run 16*(l/16)..+15
__device__ __forceinline__ v16h load_bfrag(const _Float16* row) {
  v8h lo = *(const v8h*)(row);
  v8h hi = *(const v8h*)(row + 8);
  return make_v16h(lo, hi);
}

// ---------------------------------------------------------------------------
// Kernel 1: QKV projection + meta-gating.
//   grid = (E/64, M/128, 3), block = 128 (4 waves). Each wave: 32x64 C slab.
//   Output f16 in [b][h][s][d] layout for contiguous attention fragments.
// ---------------------------------------------------------------------------
__global__ void __launch_bounds__(128)
qkv_kernel(const float* __restrict__ X,
           const float* __restrict__ Wq, const float* __restrict__ Wk,
           const float* __restrict__ Wv,
           const float* __restrict__ gq, const float* __restrict__ gk,
           _Float16* __restrict__ qo, _Float16* __restrict__ ko,
           _Float16* __restrict__ vo) {
  __shared__ _Float16 As[128*LDB];  // [m][k]
  __shared__ _Float16 Bs[64*LDB];   // [n][k]  (N-major)

  const int tid  = threadIdx.x;
  const int lane = tid & 31;
  const int wave = tid >> 5;
  const int hf   = lane >> 4;
  const int l16  = lane & 15;

  const int n0 = blockIdx.x * 64;
  const int m0 = blockIdx.y * 128;
  const int z  = blockIdx.z;

  const float* W    = (z == 0) ? Wq : ((z == 1) ? Wk : Wv);
  const float* gate = (z == 0) ? gq : ((z == 1) ? gk : nullptr);
  _Float16*    out  = (z == 0) ? qo : ((z == 1) ? ko : vo);

  const v8f vzero = {0.f,0.f,0.f,0.f,0.f,0.f,0.f,0.f};
  v8f acc0[4] = {vzero, vzero, vzero, vzero};
  v8f acc1[4] = {vzero, vzero, vzero, vzero};

  for (int k0 = 0; k0 < Ec; k0 += 64) {
    __syncthreads();
    // ---- stage A tile 128x64 (f32 -> f16), float4-vectorized, coalesced
    #pragma unroll
    for (int i = 0; i < 16; ++i) {
      int linear = tid + i*128;            // 2048 float4 chunks
      int r  = linear >> 4;                // 0..127
      int c4 = (linear & 15) << 2;         // 0..60
      const float4 f = *(const float4*)&X[(size_t)(m0 + r)*Ec + k0 + c4];
      _Float16* dst = &As[r*LDB + c4];
      dst[0]=(_Float16)f.x; dst[1]=(_Float16)f.y;
      dst[2]=(_Float16)f.z; dst[3]=(_Float16)f.w;
    }
    // ---- stage B tile transposed: Bs[n][k] = W[k0+kk][n0+n] (coalesced over n)
    #pragma unroll
    for (int i = 0; i < 32; ++i) {
      int idx = tid + i * 128;             // 0..4095
      int n   = idx & 63;
      int kk  = idx >> 6;                  // 0..63
      Bs[n*LDB + kk] = (_Float16)W[(size_t)(k0 + kk)*Ec + n0 + n];
    }
    __syncthreads();

    #pragma unroll
    for (int kk = 0; kk < 2; ++kk) {       // two 32-wide k-steps per stage
      const int ko = kk * 32;
      v16h a0 = load_afrag(&As[(wave*32      + l16)*LDB + ko], hf);
      v16h a1 = load_afrag(&As[(wave*32 + 16 + l16)*LDB + ko], hf);
      #pragma unroll
      for (int tn = 0; tn < 4; ++tn) {
        v16h bf = load_bfrag(&Bs[(tn*16 + l16)*LDB + ko + 16*hf]);
        acc0[tn] = wmma_f16(a0, bf, acc0[tn]);
        acc1[tn] = wmma_f16(a1, bf, acc1[tn]);
      }
    }
  }

  // ---- epilogue: gate (*2), f16, write [b][h][s][d]
  const int b  = m0 / Sc;                  // 128-row tile never crosses a batch
  const int hn = n0 / Dc;                  // 64-col tile == one head
  _Float16* obase = out + ((size_t)(b*Hc + hn)) * Sc * Dc;
  #pragma unroll
  for (int tn = 0; tn < 4; ++tn) {
    int n = tn*16 + l16;
    float gv = gate ? gate[b*Ec + n0 + n] * 2.0f : 1.0f;
    #pragma unroll
    for (int r = 0; r < 8; ++r) {
      int s = (m0 - b*Sc) + wave*32 + r + 8*hf;
      obase[(size_t)s * Dc + n]        = (_Float16)(acc0[tn][r] * gv);
      obase[(size_t)(s + 16) * Dc + n] = (_Float16)(acc1[tn][r] * gv);
    }
  }
}

// ---------------------------------------------------------------------------
// Kernel 2: flash attention per (b,h). block = 128 (4 waves); each wave owns a
// 16-row query tile (block covers 64 rows); waves share K/V LDS tiles.
// grid = (S/64, B*H).
// ---------------------------------------------------------------------------
__global__ void __launch_bounds__(128)
attn_kernel(const _Float16* __restrict__ Q, const _Float16* __restrict__ K,
            const _Float16* __restrict__ V, _Float16* __restrict__ AO) {
  __shared__ _Float16 Ks[32*LKS];       // [key][d]  -> transposed-B for Q.K^T
  __shared__ _Float16 Vt[64*LVT];       // [d][key]  -> transposed-B for P.V
  __shared__ _Float16 Pb[4*16*LPB];     // per-wave probability tiles [16][32]

  const int tid  = threadIdx.x;
  const int lane = tid & 31;
  const int wave = tid >> 5;
  const int hf   = lane >> 4;
  const int l16  = lane & 15;

  const int s0 = blockIdx.x * 64;
  const int bh = blockIdx.y;            // b*H + h
  const _Float16* qh = Q + (size_t)bh * Sc * Dc;
  const _Float16* kh = K + (size_t)bh * Sc * Dc;
  const _Float16* vh = V + (size_t)bh * Sc * Dc;

  // Q fragments for this wave's 16 rows (D=64 -> two k-steps), from global
  const _Float16* qrow = qh + (size_t)(s0 + wave*16 + l16) * Dc;
  v16h qa0 = load_afrag(qrow,      hf);   // K = 0..31
  v16h qa1 = load_afrag(qrow + 32, hf);   // K = 32..63

  const v8f vzero = {0.f,0.f,0.f,0.f,0.f,0.f,0.f,0.f};
  v8f o[4] = {vzero, vzero, vzero, vzero};
  float mrun[8], lrun[8];
  #pragma unroll
  for (int r = 0; r < 8; ++r) { mrun[r] = -1.0e30f; lrun[r] = 0.0f; }

  _Float16* pbw = &Pb[wave * 16 * LPB];
  const float scale = 0.125f;           // 1/sqrt(64)

  for (int j0 = 0; j0 < Sc; j0 += 32) {
    __syncthreads();                    // previous iteration's LDS reads done
    {                                   // cooperative K/V tile staging (32 keys)
      int key = tid >> 2;
      int ds  = (tid & 3) << 4;
      const _Float16* ksrc = kh + (size_t)(j0 + key) * Dc + ds;
      *(v8h*)&Ks[key*LKS + ds]     = *(const v8h*)ksrc;
      *(v8h*)&Ks[key*LKS + ds + 8] = *(const v8h*)(ksrc + 8);
      const _Float16* vsrc = vh + (size_t)(j0 + key) * Dc + ds;
      v8h v0 = *(const v8h*)vsrc;
      v8h v1 = *(const v8h*)(vsrc + 8);
      #pragma unroll
      for (int e = 0; e < 8; ++e) {
        Vt[(ds + e)     * LVT + key] = v0[e];
        Vt[(ds + 8 + e) * LVT + key] = v1[e];
      }
    }
    __syncthreads();

    // ---- scores: two 16x16 tiles (keys j0..j0+15, j0+16..j0+31)
    v8f c0 = vzero, c1 = vzero;
    #pragma unroll
    for (int kk = 0; kk < 2; ++kk) {
      v16h a = kk ? qa1 : qa0;
      v16h b0 = load_bfrag(&Ks[l16        * LKS + kk*32 + 16*hf]);
      v16h b1 = load_bfrag(&Ks[(16 + l16) * LKS + kk*32 + 16*hf]);
      c0 = wmma_f16(a, b0, c0);
      c1 = wmma_f16(a, b1, c1);
    }

    // ---- online softmax (row reductions across 16-lane groups)
    #pragma unroll
    for (int r = 0; r < 8; ++r) {
      float x0 = c0[r] * scale, x1 = c1[r] * scale;
      float mx = fmaxf(x0, x1);
      #pragma unroll
      for (int off = 1; off < 16; off <<= 1)
        mx = fmaxf(mx, __shfl_xor(mx, off, 16));
      float mnew  = fmaxf(mrun[r], mx);
      float alpha = __expf(mrun[r] - mnew);
      float p0 = __expf(x0 - mnew), p1 = __expf(x1 - mnew);
      float rs = p0 + p1;
      #pragma unroll
      for (int off = 1; off < 16; off <<= 1)
        rs += __shfl_xor(rs, off, 16);
      lrun[r] = lrun[r] * alpha + rs;
      mrun[r] = mnew;
      #pragma unroll
      for (int tn = 0; tn < 4; ++tn) o[tn][r] *= alpha;
      int m = r + 8*hf;
      pbw[m*LPB + l16]      = (_Float16)p0;
      pbw[m*LPB + 16 + l16] = (_Float16)p1;
    }
    __syncthreads();                    // P tiles visible as A-fragments

    // ---- O += P(16x32) @ V(32x64)
    v16h pa = load_afrag(&pbw[l16 * LPB], hf);
    #pragma unroll
    for (int tn = 0; tn < 4; ++tn) {
      v16h vb = load_bfrag(&Vt[(tn*16 + l16) * LVT + 16*hf]);
      o[tn] = wmma_f16(pa, vb, o[tn]);
    }
  }

  // ---- normalize and write f16 in [b][s][e] (heads concatenated)
  const int b = bh >> 4, h = bh & 15;
  #pragma unroll
  for (int tn = 0; tn < 4; ++tn)
    #pragma unroll
    for (int r = 0; r < 8; ++r) {
      int m = wave*16 + r + 8*hf;
      int d = tn*16 + l16;
      AO[((size_t)(b*Sc + s0 + m)) * Ec + h*Dc + d] = (_Float16)(o[tn][r] / lrun[r]);
    }
}

// ---------------------------------------------------------------------------
// Kernel 3: output projection + residual.  res[m,f] = AO[m,:] @ W_Out[f,:] + X[m,f]
// W_Out is [out][in] row-major == already N-major for B-fragments.
// grid = (E/64, M/128), block = 128, 2x4 register blocking per wave.
// ---------------------------------------------------------------------------
__global__ void __launch_bounds__(128)
outproj_kernel(const _Float16* __restrict__ AO, const float* __restrict__ Wout,
               const float* __restrict__ X, float* __restrict__ res) {
  __shared__ _Float16 As[128*LDB];
  __shared__ _Float16 Bs[64*LDB];

  const int tid  = threadIdx.x;
  const int lane = tid & 31;
  const int wave = tid >> 5;
  const int hf   = lane >> 4;
  const int l16  = lane & 15;

  const int n0 = blockIdx.x * 64;
  const int m0 = blockIdx.y * 128;

  const v8f vzero = {0.f,0.f,0.f,0.f,0.f,0.f,0.f,0.f};
  v8f acc0[4] = {vzero, vzero, vzero, vzero};
  v8f acc1[4] = {vzero, vzero, vzero, vzero};

  for (int k0 = 0; k0 < Ec; k0 += 64) {
    __syncthreads();
    // ---- A tile 128x64: already f16, 16-B copies
    #pragma unroll
    for (int i = 0; i < 8; ++i) {
      int linear = tid + i*128;            // 1024 v8h chunks
      int r  = linear >> 3;                // 0..127
      int cs = (linear & 7) << 3;          // 0..56
      *(v8h*)&As[r*LDB + cs] = *(const v8h*)&AO[(size_t)(m0 + r)*Ec + k0 + cs];
    }
    // ---- B tile 64x64: Wout[n][k] direct (N-major already), float4 loads
    #pragma unroll
    for (int i = 0; i < 8; ++i) {
      int linear = tid + i*128;            // 1024 float4 chunks
      int n  = linear >> 4;                // 0..63
      int c4 = (linear & 15) << 2;         // 0..60
      const float4 f = *(const float4*)&Wout[(size_t)(n0 + n)*Ec + k0 + c4];
      __builtin_prefetch(&Wout[(size_t)(n0 + n)*Ec + k0 + c4 + 64], 0, 1);
      _Float16* dst = &Bs[n*LDB + c4];
      dst[0]=(_Float16)f.x; dst[1]=(_Float16)f.y;
      dst[2]=(_Float16)f.z; dst[3]=(_Float16)f.w;
    }
    __syncthreads();

    #pragma unroll
    for (int kk = 0; kk < 2; ++kk) {
      const int ko = kk * 32;
      v16h a0 = load_afrag(&As[(wave*32      + l16)*LDB + ko], hf);
      v16h a1 = load_afrag(&As[(wave*32 + 16 + l16)*LDB + ko], hf);
      #pragma unroll
      for (int tn = 0; tn < 4; ++tn) {
        v16h bf = load_bfrag(&Bs[(tn*16 + l16)*LDB + ko + 16*hf]);
        acc0[tn] = wmma_f16(a0, bf, acc0[tn]);
        acc1[tn] = wmma_f16(a1, bf, acc1[tn]);
      }
    }
  }

  #pragma unroll
  for (int tn = 0; tn < 4; ++tn)
    #pragma unroll
    for (int r = 0; r < 8; ++r) {
      int m = wave*32 + r + 8*hf;
      int n = tn*16 + l16;
      size_t i0 = (size_t)(m0 + m) * Ec + n0 + n;
      size_t i1 = (size_t)(m0 + m + 16) * Ec + n0 + n;
      res[i0] = acc0[tn][r] + X[i0];       // residual add, f32
      res[i1] = acc1[tn][r] + X[i1];
    }
}

// ---------------------------------------------------------------------------
// Kernel 4: LayerNorm(eps=1e-6) over E=1024. One 256-thread block per row.
// ---------------------------------------------------------------------------
__global__ void __launch_bounds__(256)
ln_kernel(const float* __restrict__ res, const float* __restrict__ gamma,
          const float* __restrict__ beta, float* __restrict__ out) {
  __shared__ float red[16];
  const int row = blockIdx.x;
  const float* r = res + (size_t)row * Ec;

  float vals[4];
  float s = 0.f, s2 = 0.f;
  #pragma unroll
  for (int i = 0; i < 4; ++i) {
    float x = r[threadIdx.x + i*256];
    vals[i] = x; s += x; s2 += x*x;
  }
  #pragma unroll
  for (int off = 16; off; off >>= 1) {
    s  += __shfl_xor(s,  off, 32);
    s2 += __shfl_xor(s2, off, 32);
  }
  int w = threadIdx.x >> 5;
  if ((threadIdx.x & 31) == 0) { red[w] = s; red[8 + w] = s2; }
  __syncthreads();
  if (threadIdx.x == 0) {
    float a = 0.f, b = 0.f;
    #pragma unroll
    for (int i = 0; i < 8; ++i) { a += red[i]; b += red[8 + i]; }
    red[0] = a; red[8] = b;
  }
  __syncthreads();
  float mean = red[0] * (1.0f / Ec);
  float var  = red[8] * (1.0f / Ec) - mean * mean;
  float inv  = rsqrtf(var + 1e-6f);
  #pragma unroll
  for (int i = 0; i < 4; ++i) {
    int c = threadIdx.x + i*256;
    out[(size_t)row * Ec + c] = (vals[i] - mean) * inv * gamma[c] + beta[c];
  }
}

// ---------------------------------------------------------------------------
extern "C" void kernel_launch(void* const* d_in, const int* in_sizes, int n_in,
                              void* d_out, int out_size, void* d_ws, size_t ws_size,
                              hipStream_t stream) {
  const float* X     = (const float*)d_in[0];
  const float* gq    = (const float*)d_in[1];
  const float* gk    = (const float*)d_in[2];
  const float* Wq    = (const float*)d_in[3];
  const float* Wk    = (const float*)d_in[4];
  const float* Wv    = (const float*)d_in[5];
  const float* Wo    = (const float*)d_in[6];
  const float* gamma = (const float*)d_in[7];
  const float* beta  = (const float*)d_in[8];
  float* out = (float*)d_out;

  const size_t NH = (size_t)Bc * Hc * Sc * Dc;     // 8M elems per tensor
  _Float16* qws  = (_Float16*)d_ws;
  _Float16* kws  = qws + NH;
  _Float16* vws  = kws + NH;
  _Float16* aows = vws + NH;                       // [B*S][E] f16
  float*    resw = (float*)(aows + NH);            // [B*S][E] f32

  dim3 blk(128);
  qkv_kernel<<<dim3(Ec/64, Mc/128, 3), blk, 0, stream>>>(
      X, Wq, Wk, Wv, gq, gk, qws, kws, vws);
  attn_kernel<<<dim3(Sc/64, Bc*Hc), blk, 0, stream>>>(qws, kws, vws, aows);
  outproj_kernel<<<dim3(Ec/64, Mc/128), blk, 0, stream>>>(aows, Wo, X, resw);
  ln_kernel<<<dim3(Mc), dim3(256), 0, stream>>>(resw, gamma, beta, out);
}
